// VQVAEModel_7679401525507
// MI455X (gfx1250) — matrix operand order repaired and checked by reading, not compiled, not run on hardware
//
#include <hip/hip_runtime.h>
#include <hip/hip_bf16.h>

// ---------------------------------------------------------------------------
// VQ-VAE forward for MI455X (gfx1250, wave32).
// Heavy stages: implicit-GEMM on v_wmma_f32_16x16x32_f16, A-tiles moved by the
// Tensor Data Mover (tensor_load_to_lds + s_wait_tensorcnt) when available.
// ---------------------------------------------------------------------------

typedef __attribute__((ext_vector_type(16))) _Float16 v16h;
typedef __attribute__((ext_vector_type(8)))  _Float16 v8h;
typedef __attribute__((ext_vector_type(8)))  float    v8f;
typedef __attribute__((ext_vector_type(4)))  unsigned int u32x4;
typedef __attribute__((ext_vector_type(4)))  int          i32x4;
typedef __attribute__((ext_vector_type(8)))  int          i32x8;

#define NB   8192        // batch
#define BM   128
#define BN   64
#define BK   32
#define ASTR 40          // LDS row stride in halves (80B: 16B-aligned, conflict-free)
#define BSTR 40

#define MODE_CONV  0     // stride-2 conv, K=4, pad (1,1)
#define MODE_TCONV 1     // lhs-dilation-2 tconv, pad (2,3), weights pre-flipped
#define MODE_FCACT 2     // gather from [C][n][16] activation layout
#define MODE_ROW   3     // plain row-major K x Ncols

#if __has_builtin(__builtin_amdgcn_tensor_load_to_lds) && __has_builtin(__builtin_amdgcn_s_wait_tensorcnt)
#define HAVE_TDM 1
#else
#define HAVE_TDM 0
#endif

static __device__ __forceinline__ v8h zero8h() {
    v8h r;
#pragma unroll
    for (int i = 0; i < 8; ++i) r[i] = (_Float16)0.0f;
    return r;
}
static __device__ __forceinline__ v8f zero8f() {
    v8f r;
#pragma unroll
    for (int i = 0; i < 8; ++i) r[i] = 0.0f;
    return r;
}
static __device__ __forceinline__ v16h cat16(v8h lo, v8h hi) {
    v16h r;
#pragma unroll
    for (int i = 0; i < 8; ++i) { r[i] = lo[i]; r[i + 8] = hi[i]; }
    return r;
}

#if HAVE_TDM
// TDM descriptor (ISA 8.3/8.4): 2-D f16 tile, 128 rows x 32 halves, row stride
// K halves in memory; LDS destination padded 4 DWORDs every 16 DWORDs so the
// LDS row stride is 40 halves (== ASTR), matching the fragment loader and
// avoiding bank conflicts.  Rows beyond tensor_dim1 are zero-filled by TDM.
static __device__ __forceinline__ void tdm_load_A(const _Float16* gsrc,
                                                  unsigned lds_off,
                                                  int rows_rem, int K)
{
    const unsigned long long ga = (unsigned long long)(uintptr_t)gsrc;
    u32x4 g0;
    g0[0] = 1u;                                                  // count=1, load
    g0[1] = lds_off;                                             // lds_addr
    g0[2] = (unsigned)ga;                                        // global_addr lo
    g0[3] = (unsigned)((ga >> 32) & 0x1FFFFFFu) | 0x80000000u;   // addr hi | type=2
    i32x8 g1;
    g1[0] = (1 << 16) | (1 << 20) | (3 << 22) | (3 << 25);       // 2B elems, pad_en,
                                                                 // interval=16dw, pad=4dw
    g1[1] = (int)(((unsigned)K & 0xFFFFu) << 16);                // tensor_dim0 lo
    g1[2] = (int)(((unsigned)rows_rem & 0xFFFFu) << 16);         // tensor_dim1 lo
    g1[3] = (int)(32u << 16);                                    // tile_dim0 = 32
    g1[4] = 128;                                                 // tile_dim1 = 128
    g1[5] = K;                                                   // dim0 stride lo
    g1[6] = 0;
    g1[7] = 0;
    i32x4 z4; z4[0] = 0; z4[1] = 0; z4[2] = 0; z4[3] = 0;
    i32x8 z8;
#pragma unroll
    for (int i = 0; i < 8; ++i) z8[i] = 0;
    __builtin_amdgcn_tensor_load_to_lds(g0, g1, z4, z4, z8, 0);
}
#endif

// ---------------------------------------------------------------------------
// Implicit-GEMM:  Out[M x Ncols] = epilogue( A[M x K] * B[K x Ncols] )
// A: pre-packed f16 row-major (staged by TDM).  B: gathered from f32
// activations per MODE, converted to f16 while staging into LDS.
// 8 waves: 4(M) x 2(N); each wave owns 32x32 = 2x2 WMMA 16x16x32 accumulators.
// ---------------------------------------------------------------------------
template <int MODE, int RELU>
__global__ __launch_bounds__(256) void gemm_wmma(
    const _Float16* __restrict__ A, const float* __restrict__ Bsrc,
    float* __restrict__ Out,
    const float* __restrict__ gvec, const float* __restrict__ bvec,
    int M, int K, int Ncols, int Lin, int Lout, int inNL, float scale_const)
{
    __shared__ __align__(16) _Float16 As[2][BM * ASTR];
    __shared__ __align__(16) _Float16 Bs[2][BN * BSTR];

    const int tid   = threadIdx.x;
    const int lane  = tid & 31;
    const int wid   = tid >> 5;
    const int wm    = wid >> 1;            // 0..3
    const int wn    = wid & 1;             // 0..1
    const int mrow0 = blockIdx.y * BM;
    const int ncol0 = blockIdx.x * BN;

    // B gather coordinates: one column, 8 consecutive k per thread
    const int bcol  = tid & 63;
    const int bkseg = tid >> 6;            // 0..3
    const int gcol  = ncol0 + bcol;
    int bn = gcol, bl = 0;
    if (MODE == MODE_CONV || MODE == MODE_TCONV) { bn = gcol / Lout; bl = gcol - bn * Lout; }

#if HAVE_TDM
    const unsigned ldsA[2] = { (unsigned)(uintptr_t)&As[0][0],
                               (unsigned)(uintptr_t)&As[1][0] };
    auto stageA = [&](int kb, int b) {
        if (wid == 0)
            tdm_load_A(A + (size_t)mrow0 * K + kb * BK, ldsA[b], M - mrow0, K);
    };
#else
    const int  arow = tid >> 1;
    const int  aseg = tid & 1;
    const bool a_ok = (mrow0 + arow) < M;
    const _Float16* aptr = A + (size_t)(mrow0 + arow) * K + aseg * 16;
    auto stageA = [&](int kb, int b) {
        v8h x0 = zero8h(), x1 = zero8h();
        if (a_ok) {
            const _Float16* p = aptr + kb * BK;
            x0 = *(const v8h*)(p);
            x1 = *(const v8h*)(p + 8);
            __builtin_prefetch((const void*)(p + BK), 0, 0);
        }
        *(v8h*)&As[b][arow * ASTR + aseg * 16]     = x0;
        *(v8h*)&As[b][arow * ASTR + aseg * 16 + 8] = x1;
    };
#endif

    auto stageB = [&](int kb, int b) {
        v8h t = zero8h();
        if (MODE == MODE_FCACT) {
            // 8 consecutive k's are 8 consecutive floats: two b128 loads
            const int kk0 = kb * BK + bkseg * 8;
            const int c = kk0 >> 4, pos0 = kk0 & 15;
            const float* src = Bsrc + (size_t)c * inNL + (size_t)gcol * 16 + pos0;
            const float4 f0 = *(const float4*)(src);
            const float4 f1 = *(const float4*)(src + 4);
            t[0] = (_Float16)f0.x; t[1] = (_Float16)f0.y;
            t[2] = (_Float16)f0.z; t[3] = (_Float16)f0.w;
            t[4] = (_Float16)f1.x; t[5] = (_Float16)f1.y;
            t[6] = (_Float16)f1.z; t[7] = (_Float16)f1.w;
        } else {
#pragma unroll
            for (int j = 0; j < 8; ++j) {
                const int kk = kb * BK + bkseg * 8 + j;
                float v = 0.0f;
                if (MODE == MODE_CONV) {
                    const int ci = kk >> 2, tp = kk & 3;
                    const int p  = 2 * bl - 1 + tp;
                    if (p >= 0 && p < Lin)
                        v = Bsrc[(size_t)ci * inNL + (size_t)bn * Lin + p];
                } else if (MODE == MODE_TCONV) {
                    const int ci = kk >> 2, tp = kk & 3;
                    const int p  = bl - 2 + tp;                  // dilated coordinate
                    if (p >= 0 && p <= 2 * Lin - 2 && ((p & 1) == 0))
                        v = Bsrc[(size_t)ci * inNL + (size_t)bn * Lin + (p >> 1)];
                } else {                                         // MODE_ROW
                    v = Bsrc[(size_t)kk * Ncols + gcol];
                }
                t[j] = (_Float16)v;
            }
        }
        *(v8h*)&Bs[b][bcol * BSTR + bkseg * 8] = t;              // transposed: Bs[n][k]
    };

    v8f acc[2][2];
#pragma unroll
    for (int i = 0; i < 2; ++i)
#pragma unroll
        for (int j = 0; j < 2; ++j) acc[i][j] = zero8f();

    const int lr   = lane & 15;
    const int asel = (lane < 16) ? 0 : 8;    // A: K {0-7,16-23} vs {8-15,24-31}
    const int bsel = (lane < 16) ? 0 : 8;    // B: K {0-7,16-23} vs {8-15,24-31}

    const int nk = K / BK;
    stageA(0, 0);
    stageB(0, 0);
#if HAVE_TDM
    if (wid == 0) __builtin_amdgcn_s_wait_tensorcnt(0);
#endif
    __syncthreads();

    int buf = 0;
    for (int kb = 0; kb < nk; ++kb) {
        if (kb + 1 < nk) { stageA(kb + 1, buf ^ 1); stageB(kb + 1, buf ^ 1); }

        v16h af[2], bf[2];
#pragma unroll
        for (int s = 0; s < 2; ++s) {
            const _Float16* ap = &As[buf][(wm * 32 + s * 16 + lr) * ASTR];
            v8h alo = *(const v8h*)(ap + asel);
            v8h ahi = *(const v8h*)(ap + asel + 16);
            af[s] = cat16(alo, ahi);
            const _Float16* bp = &Bs[buf][(wn * 32 + s * 16 + lr) * BSTR + bsel];
            v8h blo = *(const v8h*)(bp);
            v8h bhi = *(const v8h*)(bp + 16);
            bf[s] = cat16(blo, bhi);
        }
#pragma unroll
        for (int i = 0; i < 2; ++i)
#pragma unroll
            for (int j = 0; j < 2; ++j)
                acc[i][j] = __builtin_amdgcn_wmma_f32_16x16x32_f16(
                    false, af[i], false, bf[j], (short)0, acc[i][j], false, false);

#if HAVE_TDM
        if (wid == 0) __builtin_amdgcn_s_wait_tensorcnt(0);
#endif
        __syncthreads();
        buf ^= 1;
    }

    // Epilogue: out = opt_relu(scale_const * g[row] * acc + b[row])
#pragma unroll
    for (int i = 0; i < 2; ++i) {
#pragma unroll
        for (int j = 0; j < 2; ++j) {
            const int colg = ncol0 + wn * 32 + j * 16 + lr;
#pragma unroll
            for (int r = 0; r < 8; ++r) {
                const int rowg = mrow0 + wm * 32 + i * 16 + ((lane < 16) ? 0 : 8) + r;
                if (rowg < M) {
                    const float g  = gvec ? gvec[rowg] : 1.0f;
                    const float bb = bvec ? bvec[rowg] : 0.0f;
                    float v = scale_const * g * acc[i][j][r] + bb;
                    if (RELU) v = fmaxf(v, 0.0f);
                    Out[(size_t)rowg * Ncols + colg] = v;
                }
            }
        }
    }
}

// --------------------------- small helper kernels ---------------------------

__global__ void pack_w_kernel(const float* __restrict__ w, _Float16* __restrict__ o,
                              int total, int K, int flip)
{
    int t = blockIdx.x * 256 + threadIdx.x;
    if (t >= total) return;
    int k  = t % K;
    int kt = flip ? ((k & ~3) | (3 - (k & 3))) : k;
    o[t - k + kt] = (_Float16)w[t];
}

__global__ void cnorm_kernel(const float* __restrict__ cb, float* __restrict__ cn)
{
    int j = blockIdx.x * 256 + threadIdx.x;
    if (j >= 1024) return;
    float s = 0.0f;
    for (int c = 0; c < 128; ++c) { float v = cb[j * 128 + c]; s += v * v; }
    cn[j] = s;
}

// Encoder conv0: Cin=1, Cout=32, L 256->128 ; out layout [c][n][128]
__global__ __launch_bounds__(256) void enc0_kernel(
    const float* __restrict__ x, const float* __restrict__ w,
    const float* __restrict__ g, const float* __restrict__ be,
    float* __restrict__ out)
{
    int tid = blockIdx.x * 256 + threadIdx.x;   // n*128 + l
    int n = tid >> 7, l = tid & 127;
    float in4[4];
#pragma unroll
    for (int k = 0; k < 4; ++k) {
        int p = 2 * l - 1 + k;
        in4[k] = (p >= 0 && p < 256) ? x[(size_t)n * 256 + p] : 0.0f;
    }
    for (int c = 0; c < 32; ++c) {
        float s = 0.0f;
#pragma unroll
        for (int k = 0; k < 4; ++k) s += w[c * 4 + k] * in4[k];
        out[(size_t)c * (NB * 128) + tid] = fmaxf(g[c] * s + be[c], 0.0f);
    }
}

__global__ void argmin_kernel(const float* __restrict__ sc, int* __restrict__ idxw,
                              float* __restrict__ outIdx)
{
    int n = blockIdx.x * 256 + threadIdx.x;
    if (n >= NB) return;
    float best = sc[n]; int bi = 0;
    for (int j = 1; j < 1024; ++j) {
        float v = sc[(size_t)j * NB + n];
        if (v < best) { best = v; bi = j; }
    }
    idxw[n] = bi;
    outIdx[n] = (float)bi;
}

__global__ void init_accum_kernel(float* a) { a[0] = 0.0f; }

__global__ __launch_bounds__(256) void zq_loss_kernel(
    const int* __restrict__ idx, const float* __restrict__ cb,
    const float* __restrict__ ze, float* __restrict__ zq, float* __restrict__ accum)
{
    __shared__ float red[256];
    int tid = blockIdx.x * 256 + threadIdx.x;   // c*8192 + n, c<128
    int c = tid >> 13, n = tid & (NB - 1);
    float q = cb[(size_t)idx[n] * 128 + c];
    float z = ze[(size_t)c * NB + n];
    zq[(size_t)c * NB + n] = q;
    float d = q - z;
    red[threadIdx.x] = d * d;
    __syncthreads();
    for (int off = 128; off > 0; off >>= 1) {
        if (threadIdx.x < off) red[threadIdx.x] += red[threadIdx.x + off];
        __syncthreads();
    }
    if (threadIdx.x == 0) atomicAdd(accum, red[0]);
}

__global__ void finalize_kernel(const float* __restrict__ accum, float* __restrict__ out2)
{
    float m = accum[0] * (1.0f / ((float)NB * 128.0f));
    out2[0] = m;   // commitment_loss
    out2[1] = m;   // codebook_loss (same forward value)
}

// [4096][n] FC output -> [c=256][n][16] activation layout
__global__ void repack_kernel(const float* __restrict__ H, float* __restrict__ O)
{
    size_t t = (size_t)blockIdx.x * 256 + threadIdx.x;  // f*8192 + n
    int f = (int)(t >> 13), n = (int)(t & (NB - 1));
    int c = f >> 4, pos = f & 15;
    O[(size_t)c * (NB * 16) + (size_t)n * 16 + pos] = H[t];
}

// Final tconv (Cout=1) cropped to 256 + row softmax.  in: [32][n][135]
__global__ __launch_bounds__(256) void dec3_softmax_kernel(
    const float* __restrict__ in, const float* __restrict__ w3,
    const float* __restrict__ b3, float* __restrict__ out)
{
    __shared__ float wf[128];
    __shared__ float red[256];
    const int n = blockIdx.x;
    const int t = threadIdx.x;
    if (t < 128) wf[t] = w3[t];
    __syncthreads();

    float s = b3[0];
#pragma unroll
    for (int k = 0; k < 4; ++k) {
        const int p = t - 2 + k;                 // dilated coordinate
        if (p >= 0 && p <= 268 && ((p & 1) == 0)) {
            const int pos = p >> 1;
            for (int ci = 0; ci < 32; ++ci)
                s += in[(size_t)ci * (NB * 135) + (size_t)n * 135 + pos] * wf[ci * 4 + (3 - k)];
        }
    }
    red[t] = s;
    __syncthreads();
    for (int off = 128; off > 0; off >>= 1) {
        if (t < off) red[t] = fmaxf(red[t], red[t + off]);
        __syncthreads();
    }
    const float mx = red[0];
    __syncthreads();
    const float e = __expf(s - mx);
    red[t] = e;
    __syncthreads();
    for (int off = 128; off > 0; off >>= 1) {
        if (t < off) red[t] += red[t + off];
        __syncthreads();
    }
    out[(size_t)n * 256 + t] = e * (1.0f / red[0]);
}

// ---------------------------------------------------------------------------

extern "C" void kernel_launch(void* const* d_in, const int* in_sizes, int n_in,
                              void* d_out, int out_size, void* d_ws, size_t ws_size,
                              hipStream_t stream)
{
    (void)in_sizes; (void)n_in; (void)out_size; (void)ws_size;

    const float* x        = (const float*)d_in[0];
    const float* enc_w0   = (const float*)d_in[1];
    const float* enc_g0   = (const float*)d_in[2];
    const float* enc_be0  = (const float*)d_in[3];
    const float* enc_w1   = (const float*)d_in[4];
    const float* enc_g1   = (const float*)d_in[5];
    const float* enc_be1  = (const float*)d_in[6];
    const float* enc_w2   = (const float*)d_in[7];
    const float* enc_g2   = (const float*)d_in[8];
    const float* enc_be2  = (const float*)d_in[9];
    const float* enc_w3   = (const float*)d_in[10];
    const float* enc_g3   = (const float*)d_in[11];
    const float* enc_be3  = (const float*)d_in[12];
    const float* enc_fc_w = (const float*)d_in[13];
    const float* enc_fc_b = (const float*)d_in[14];
    const float* codebook = (const float*)d_in[15];
    const float* dec_fc_w = (const float*)d_in[16];
    const float* dec_fc_b = (const float*)d_in[17];
    const float* dec_w0   = (const float*)d_in[18];
    const float* dec_g0   = (const float*)d_in[19];
    const float* dec_be0  = (const float*)d_in[20];
    const float* dec_w1   = (const float*)d_in[21];
    const float* dec_g1   = (const float*)d_in[22];
    const float* dec_be1  = (const float*)d_in[23];
    const float* dec_w2   = (const float*)d_in[24];
    const float* dec_g2   = (const float*)d_in[25];
    const float* dec_be2  = (const float*)d_in[26];
    const float* dec_w3   = (const float*)d_in[27];
    const float* dec_b3   = (const float*)d_in[28];

    float* out     = (float*)d_out;
    float* out_sc  = out + (size_t)NB * 256;       // [commitment, codebook]
    float* out_idx = out + (size_t)NB * 256 + 2;   // idx as float

    // ---- workspace carve-out ----
    char*  base = (char*)d_ws;
    size_t cur  = 0;
    auto alloc = [&](size_t bytes) -> char* {
        char* p = base + cur;
        cur += (bytes + 255) & ~(size_t)255;
        return p;
    };
    const size_t SZP = 142606336;                   // >= 32*8192*135*4
    float*     P0      = (float*)alloc(SZP);
    float*     P1      = (float*)alloc(SZP);
    float*     ze      = (float*)alloc((size_t)128 * NB * 4);
    float*     zq      = (float*)alloc((size_t)128 * NB * 4);
    float*     scores  = (float*)alloc((size_t)1024 * NB * 4);
    int*       idxw    = (int*)  alloc((size_t)NB * 4);
    float*     accum   = (float*)alloc(256);
    float*     cnv     = (float*)alloc(1024 * 4);
    _Float16*  wp_e1   = (_Float16*)alloc((size_t)64  * 128  * 2);
    _Float16*  wp_e2   = (_Float16*)alloc((size_t)128 * 256  * 2);
    _Float16*  wp_e3   = (_Float16*)alloc((size_t)256 * 512  * 2);
    _Float16*  wp_fc   = (_Float16*)alloc((size_t)128 * 4096 * 2);
    _Float16*  wp_cb   = (_Float16*)alloc((size_t)1024 * 128 * 2);
    _Float16*  wp_dfc  = (_Float16*)alloc((size_t)4096 * 128 * 2);
    _Float16*  wp_d0   = (_Float16*)alloc((size_t)128 * 1024 * 2);
    _Float16*  wp_d1   = (_Float16*)alloc((size_t)64  * 512  * 2);
    _Float16*  wp_d2   = (_Float16*)alloc((size_t)32  * 256  * 2);

    auto pack = [&](const float* w, _Float16* o, int M, int K, int flip) {
        int total = M * K;
        pack_w_kernel<<<(total + 255) / 256, 256, 0, stream>>>(w, o, total, K, flip);
    };
    pack(enc_w1,   wp_e1,  64,  128,  0);
    pack(enc_w2,   wp_e2,  128, 256,  0);
    pack(enc_w3,   wp_e3,  256, 512,  0);
    pack(enc_fc_w, wp_fc,  128, 4096, 0);
    pack(codebook, wp_cb,  1024, 128, 0);
    pack(dec_fc_w, wp_dfc, 4096, 128, 0);
    pack(dec_w0,   wp_d0,  128, 1024, 1);
    pack(dec_w1,   wp_d1,  64,  512,  1);
    pack(dec_w2,   wp_d2,  32,  256,  1);
    cnorm_kernel<<<4, 256, 0, stream>>>(codebook, cnv);

    auto gemm = [&](const _Float16* A, const float* B, float* O,
                    const float* g, const float* b,
                    int M, int K, int Ncols, int mode, int Lin, int Lout, int inNL,
                    float sc) {
        dim3 grid(Ncols / BN, (M + BM - 1) / BM);
        switch (mode) {
        case MODE_CONV:
            gemm_wmma<MODE_CONV, 1><<<grid, 256, 0, stream>>>(A, B, O, g, b, M, K, Ncols, Lin, Lout, inNL, sc);
            break;
        case MODE_TCONV:
            gemm_wmma<MODE_TCONV, 1><<<grid, 256, 0, stream>>>(A, B, O, g, b, M, K, Ncols, Lin, Lout, inNL, sc);
            break;
        case MODE_FCACT:
            gemm_wmma<MODE_FCACT, 0><<<grid, 256, 0, stream>>>(A, B, O, g, b, M, K, Ncols, Lin, Lout, inNL, sc);
            break;
        default:
            gemm_wmma<MODE_ROW, 0><<<grid, 256, 0, stream>>>(A, B, O, g, b, M, K, Ncols, Lin, Lout, inNL, sc);
            break;
        }
    };

    // ---- encoder ----
    enc0_kernel<<<(NB * 128) / 256, 256, 0, stream>>>(x, enc_w0, enc_g0, enc_be0, P0);
    gemm(wp_e1, P0, P1, enc_g1, enc_be1, 64,  128, NB * 64, MODE_CONV, 128, 64, NB * 128, 1.0f);
    gemm(wp_e2, P1, P0, enc_g2, enc_be2, 128, 256, NB * 32, MODE_CONV, 64,  32, NB * 64,  1.0f);
    gemm(wp_e3, P0, P1, enc_g3, enc_be3, 256, 512, NB * 16, MODE_CONV, 32,  16, NB * 32,  1.0f);
    gemm(wp_fc, P1, ze, nullptr, enc_fc_b, 128, 4096, NB, MODE_FCACT, 16, 0, NB * 16, 1.0f);

    // ---- vector quantization:  score = ||c||^2 - 2 <z_e, c>  (argmin-equivalent) ----
    gemm(wp_cb, ze, scores, nullptr, cnv, 1024, 128, NB, MODE_ROW, 0, 0, 0, -2.0f);
    argmin_kernel<<<NB / 256, 256, 0, stream>>>(scores, idxw, out_idx);
    init_accum_kernel<<<1, 1, 0, stream>>>(accum);
    zq_loss_kernel<<<(128 * NB) / 256, 256, 0, stream>>>(idxw, codebook, ze, zq, accum);
    finalize_kernel<<<1, 1, 0, stream>>>(accum, out_sc);

    // ---- decoder ----
    gemm(wp_dfc, zq, P0, nullptr, dec_fc_b, 4096, 128, NB, MODE_ROW, 0, 0, 0, 1.0f);
    repack_kernel<<<(4096 * NB) / 256, 256, 0, stream>>>(P0, P1);   // -> [256][n][16]
    gemm(wp_d0, P1, P0, dec_g0, dec_be0, 128, 1024, NB * 33,  MODE_TCONV, 16, 33,  NB * 16, 1.0f);
    gemm(wp_d1, P0, P1, dec_g1, dec_be1, 64,  512,  NB * 67,  MODE_TCONV, 33, 67,  NB * 33, 1.0f);
    gemm(wp_d2, P1, P0, dec_g2, dec_be2, 32,  256,  NB * 135, MODE_TCONV, 67, 135, NB * 67, 1.0f);
    dec3_softmax_kernel<<<NB, 256, 0, stream>>>(P0, dec_w3, dec_b3, out);
}
